// DecoderBlock_14413910245790
// MI455X (gfx1250) — matrix operand order repaired and checked by reading, not compiled, not run on hardware
//
#include <hip/hip_runtime.h>

// ---------------------------------------------------------------------------
// MI455X (gfx1250) decoder block: bf16 WMMA everywhere, f32 accumulate.
// ~450 GFLOP vs ~30us of HBM traffic at 23.3 TB/s -> matrix-core bound;
// V_WMMA_F32_16X16X32_BF16 is the throughput path (f32 WMMA is 16x16x4 only).
// ---------------------------------------------------------------------------

typedef __bf16 bf16;
typedef bf16 v16bf __attribute__((ext_vector_type(16)));
typedef bf16 v8bf  __attribute__((ext_vector_type(8)));
typedef float v8f  __attribute__((ext_vector_type(8)));
typedef int   v4i32 __attribute__((ext_vector_type(4)));

union BF16x16 { v16bf v; v8bf h[2]; };

// ---- CDNA5 async global->LDS copy (guarded; sync fallback) ----------------
#if defined(__has_builtin)
#if __has_builtin(__builtin_amdgcn_global_load_async_to_lds_b128)
#define HAVE_ASYNC_COPY 1
#endif
#endif

typedef __attribute__((address_space(1))) v4i32* gptr_v4i;
typedef __attribute__((address_space(3))) v4i32* lptr_v4i;

__device__ __forceinline__ void cp16(const void* g, void* l) {
#ifdef HAVE_ASYNC_COPY
  __builtin_amdgcn_global_load_async_to_lds_b128(
      (gptr_v4i)(v4i32*)g, (lptr_v4i)(v4i32*)l, 0, 0);
#else
  *(uint4*)l = *(const uint4*)g;
#endif
}

__device__ __forceinline__ void async_wait0() {
#ifdef HAVE_ASYNC_COPY
#if __has_builtin(__builtin_amdgcn_s_wait_asynccnt)
  __builtin_amdgcn_s_wait_asynccnt(0);
#else
  asm volatile("s_wait_asynccnt 0" ::: "memory");
#endif
#endif
}

__device__ __forceinline__ void lds_wave_fence() {
  // DS ops from one wave are in-order (ISA 7.3); just stop compiler reordering
  // and drain DScnt for safety.
#if defined(__has_builtin) && __has_builtin(__builtin_amdgcn_s_wait_dscnt)
  asm volatile("" ::: "memory");
  __builtin_amdgcn_s_wait_dscnt(0);
  asm volatile("" ::: "memory");
#else
  asm volatile("s_wait_dscnt 0" ::: "memory");
#endif
}

// ---- WMMA helpers ---------------------------------------------------------
__device__ __forceinline__ v8f wmma_bf16(const BF16x16& a, const BF16x16& b, v8f c) {
  return __builtin_amdgcn_wmma_f32_16x16x32_bf16(false, a.v, false, b.v,
                                                 (short)0, c, false, false);
}

// A fragment, 16x32 bf16, row-major source with given element stride.
// Lane L: row = L&15; halves 0-7 -> K=koff..koff+7, halves 8-15 -> K=koff+16..+23
// where koff = (L>>4)*8.  (ISA 7.12.2, 16-bit A matrix 16x32)
__device__ __forceinline__ BF16x16 load_frag_A(const bf16* base, int stride, int lane) {
  int row = lane & 15, ko = (lane >> 4) * 8;
  BF16x16 f;
  f.h[0] = *(const v8bf*)(base + row * stride + ko);
  f.h[1] = *(const v8bf*)(base + row * stride + ko + 16);
  return f;
}

// B fragment, 32x16 bf16, from N-major storage (row n holds K contiguously).
// Lane L: col n = L&15; halves 0-7 -> K=koff..koff+7, halves 8-15 -> K=koff+16..+23, koff = (L>>4)*8.
__device__ __forceinline__ BF16x16 load_frag_B(const bf16* base, int stride, int lane) {
  int n = lane & 15, ko = (lane >> 4) * 8;
  BF16x16 f;
  f.h[0] = *(const v8bf*)(base + n * stride + ko);
  f.h[1] = *(const v8bf*)(base + n * stride + ko + 16);
  return f;
}

// ---------------------------------------------------------------------------
// LayerNorm (optionally fused residual add): x [rows][2048] -> bf16 out.
// ---------------------------------------------------------------------------
template<bool RES>
__global__ __launch_bounds__(256) void ln_kernel(
    const float* __restrict__ x, const float* __restrict__ add,
    const float* __restrict__ scale, const float* __restrict__ bias,
    float* __restrict__ x1out, bf16* __restrict__ r) {
  const int D = 2048;
  long base = (long)blockIdx.x * D;
  int tid = threadIdx.x;
  float vals[8], sum = 0.f, sq = 0.f;
#pragma unroll
  for (int i = 0; i < 8; ++i) {
    int idx = tid + i * 256;
    float v = x[base + idx];
    if (RES) { v += add[base + idx]; x1out[base + idx] = v; }
    vals[i] = v; sum += v; sq += v * v;
  }
  __shared__ float s1[256], s2[256];
  s1[tid] = sum; s2[tid] = sq;
  __syncthreads();
  for (int off = 128; off > 0; off >>= 1) {
    if (tid < off) { s1[tid] += s1[tid + off]; s2[tid] += s2[tid + off]; }
    __syncthreads();
  }
  float mu = s1[0] * (1.f / D);
  float var = s2[0] * (1.f / D) - mu * mu;
  float rs = rsqrtf(var + 1e-6f);
#pragma unroll
  for (int i = 0; i < 8; ++i) {
    int idx = tid + i * 256;
    r[base + idx] = (bf16)((vals[i] - mu) * rs * scale[idx] + bias[idx]);
  }
}

// ---------------------------------------------------------------------------
// Weight transpose + fp32->bf16: W[K][N] -> Wt[N][K] (N-major for B frags).
// ---------------------------------------------------------------------------
__global__ __launch_bounds__(256) void convt_kernel(
    const float* __restrict__ W, bf16* __restrict__ Wt, int Kd, int Nd) {
  __shared__ float t[32][33];
  int n0 = blockIdx.x * 32, k0 = blockIdx.y * 32;
  int tx = threadIdx.x, ty = threadIdx.y;
#pragma unroll
  for (int i = 0; i < 4; ++i)
    t[ty * 4 + i][tx] = W[(long)(k0 + ty * 4 + i) * Nd + n0 + tx];
  __syncthreads();
#pragma unroll
  for (int i = 0; i < 4; ++i)
    Wt[(long)(n0 + ty * 4 + i) * Kd + k0 + tx] = (bf16)t[tx][ty * 4 + i];
}

// ---------------------------------------------------------------------------
// bf16 WMMA GEMM: C[M][N] = A[M][K] * Bt[N][K]^T + bias, fused epilogue.
// Block tile 128x128, 8 waves (4 along M x 2 along N), wave tile 32x64,
// K step 32, double-buffered LDS fed by async copies.
// MODE 0: +bias -> bf16   (QKV)
// MODE 1: +bias, tanh-GELU -> bf16 (MLP up)
// MODE 2: +bias +resid -> f32 (MLP down + residual)
// ---------------------------------------------------------------------------
template<int MODE>
__global__ __launch_bounds__(256) void gemm_bf16_wmma(
    const bf16* __restrict__ A, const bf16* __restrict__ Bt,
    const float* __restrict__ bias, bf16* __restrict__ outb,
    float* __restrict__ outf, const float* __restrict__ resid,
    int M, int N, int K) {
  __shared__ bf16 As[2][128 * 40];
  __shared__ bf16 Bs[2][128 * 40];
  int tid = threadIdx.x, lane = tid & 31, wave = tid >> 5;
  int bm = blockIdx.y * 128, bn = blockIdx.x * 128;
  int wm = (wave & 3) * 32;   // wave's M offset in tile
  int wn = (wave >> 2) * 64;  // wave's N offset in tile
  v8f acc[2][4] = {};

  auto issue_tile = [&](int kt, int buf) {
    const bf16* Ag = A  + (long)bm * K + (long)kt * 32;
    const bf16* Bg = Bt + (long)bn * K + (long)kt * 32;
#pragma unroll
    for (int i = 0; i < 2; ++i) {
      int c = (tid << 1) | i;        // 0..511 chunks of 8 halves
      int row = c >> 2, cc = c & 3;  // 128 rows x 4 chunks
      cp16(Ag + (long)row * K + cc * 8, &As[buf][row * 40 + cc * 8]);
      cp16(Bg + (long)row * K + cc * 8, &Bs[buf][row * 40 + cc * 8]);
    }
  };

  int KT = K >> 5;
  issue_tile(0, 0);
  async_wait0();
  __syncthreads();
  for (int kt = 0; kt < KT; ++kt) {
    int buf = kt & 1;
    if (kt + 1 < KT) issue_tile(kt + 1, buf ^ 1);  // overlap with compute
    const bf16* as = As[buf];
    const bf16* bs = Bs[buf];
    BF16x16 aF0 = load_frag_A(as + (wm +  0) * 40, 40, lane);
    BF16x16 aF1 = load_frag_A(as + (wm + 16) * 40, 40, lane);
#pragma unroll
    for (int nt = 0; nt < 4; ++nt) {
      BF16x16 bF = load_frag_B(bs + (wn + nt * 16) * 40, 40, lane);
      acc[0][nt] = wmma_bf16(aF0, bF, acc[0][nt]);
      acc[1][nt] = wmma_bf16(aF1, bF, acc[1][nt]);
    }
    async_wait0();
    __syncthreads();
  }

  // Epilogue. C layout: lane holds col = lane&15, rows (lane>>4)*8 + r.
#pragma unroll
  for (int mt = 0; mt < 2; ++mt) {
#pragma unroll
    for (int nt = 0; nt < 4; ++nt) {
      int col = bn + wn + nt * 16 + (lane & 15);
      int r0  = bm + wm + mt * 16 + ((lane >> 4) * 8);
      float bc = bias[col];
#pragma unroll
      for (int r = 0; r < 8; ++r) {
        long idx = (long)(r0 + r) * N + col;
        float v = acc[mt][nt][r] + bc;
        if constexpr (MODE == 0) {
          outb[idx] = (bf16)v;
        } else if constexpr (MODE == 1) {
          v = 0.5f * v * (1.f + tanhf(0.7978845608f * (v + 0.044715f * v * v * v)));
          outb[idx] = (bf16)v;
        } else {
          outf[idx] = v + resid[idx];
        }
      }
    }
  }
}

// ---------------------------------------------------------------------------
// Flash attention, causal. qkv bf16 [B*T][6144] (q|k|v each D=2048).
// Block = (b,h, 128 query rows); 8 waves x 16 rows each. Key tiles of 32.
// ---------------------------------------------------------------------------
__global__ __launch_bounds__(256) void attn_kernel(
    const bf16* __restrict__ qkv, float* __restrict__ out) {
  const int T = 2048, D = 2048, QKVS = 6144;
  const float SCL = 0.0883883476f;      // 1/sqrt(128)
  const float L2E = 1.44269504089f;

  int b = blockIdx.y >> 4, h = blockIdx.y & 15;
  int tid = threadIdx.x, lane = tid & 31, wave = tid >> 5;
  const bf16* qb = qkv + (long)b * T * QKVS + h * 128;
  const bf16* kb = qb + 2048;
  const bf16* vb = qb + 4096;

  __shared__ bf16 Ks[32 * 136];   // K tile, key-major [32][128 + pad]
  __shared__ bf16 Vt[128 * 40];   // V tile transposed, d-major [128][32 + pad]
  __shared__ bf16 Ps[8 * 16 * 40];// per-wave P scratch [16][32 + pad]

  int q0 = blockIdx.x * 128 + wave * 16;

  // Q fragments resident: 4 k-slices of 32 (DH=128)
  BF16x16 qF[4];
#pragma unroll
  for (int ds = 0; ds < 4; ++ds)
    qF[ds] = load_frag_A(qb + (long)q0 * QKVS + ds * 32, QKVS, lane);

  v8f o[8] = {};
  float mrow[8], lrow[8];
#pragma unroll
  for (int r = 0; r < 8; ++r) { mrow[r] = -1e38f; lrow[r] = 0.f; }

  int rbase = (lane >> 4) * 8, nlo = lane & 15;
  int jmax_blk = (blockIdx.x * 128 + 127) >> 5;
  int jmax_wav = (q0 + 15) >> 5;

  for (int j = 0; j <= jmax_blk; ++j) {
    __syncthreads();  // everyone done reading previous K/V tile
    // cooperative load: K tile row-major, V tile transposed
#pragma unroll
    for (int i = 0; i < 2; ++i) {
      int c = (tid << 1) | i;           // 512 chunks of 8 halves
      int row = c >> 4, cc = c & 15;
      *(v8bf*)(Ks + row * 136 + cc * 8) =
          *(const v8bf*)(kb + (long)(j * 32 + row) * QKVS + cc * 8);
      v8bf vv = *(const v8bf*)(vb + (long)(j * 32 + row) * QKVS + cc * 8);
#pragma unroll
      for (int e = 0; e < 8; ++e) Vt[(cc * 8 + e) * 40 + row] = vv[e];
    }
    __syncthreads();

    if (j <= jmax_wav) {
      // S = Q K^T : two 16x16 tiles over the 32 keys
      v8f s0 = {}, s1 = {};
#pragma unroll
      for (int ds = 0; ds < 4; ++ds) {
        BF16x16 b0 = load_frag_B(Ks + ds * 32,            136, lane);
        BF16x16 b1 = load_frag_B(Ks + 16 * 136 + ds * 32, 136, lane);
        s0 = wmma_bf16(qF[ds], b0, s0);
        s1 = wmma_bf16(qF[ds], b1, s1);
      }
      // online softmax (per-row, rows rbase..rbase+7 in this lane half)
      float corr[8];
      bf16* ps = Ps + wave * 640;
#pragma unroll
      for (int r = 0; r < 8; ++r) {
        int qr = q0 + rbase + r;
        float a = s0[r] * SCL, bb = s1[r] * SCL;
        if (j * 32 + nlo > qr)      a  = -1e38f;
        if (j * 32 + 16 + nlo > qr) bb = -1e38f;
        float mx = fmaxf(a, bb);
#pragma unroll
        for (int m = 1; m < 16; m <<= 1) mx = fmaxf(mx, __shfl_xor(mx, m, 32));
        float mo = mrow[r], mn = fmaxf(mo, mx);
        float c = exp2f((mo - mn) * L2E);
        float pa = exp2f((a - mn) * L2E), pb = exp2f((bb - mn) * L2E);
        float rs = pa + pb;
#pragma unroll
        for (int m = 1; m < 16; m <<= 1) rs += __shfl_xor(rs, m, 32);
        mrow[r] = mn; lrow[r] = lrow[r] * c + rs; corr[r] = c;
        ps[(rbase + r) * 40 + nlo]      = (bf16)pa;
        ps[(rbase + r) * 40 + 16 + nlo] = (bf16)pb;
      }
      lds_wave_fence();  // wave-local P round trip through LDS
      // O = O*corr + P V
      BF16x16 pF = load_frag_A(ps, 40, lane);
#pragma unroll
      for (int dt = 0; dt < 8; ++dt) {
        BF16x16 vF = load_frag_B(Vt + dt * 16 * 40, 40, lane);
        v8f oo = o[dt];
#pragma unroll
        for (int r = 0; r < 8; ++r) oo[r] *= corr[r];
        o[dt] = wmma_bf16(pF, vF, oo);
      }
    }
  }

  // write O / l
#pragma unroll
  for (int dt = 0; dt < 8; ++dt) {
    int col = h * 128 + dt * 16 + nlo;
#pragma unroll
    for (int r = 0; r < 8; ++r) {
      int row = q0 + rbase + r;
      out[((long)b * T + row) * D + col] = o[dt][r] / lrow[r];
    }
  }
}

// ---------------------------------------------------------------------------
extern "C" void kernel_launch(void* const* d_in, const int* in_sizes, int n_in,
                              void* d_out, int out_size, void* d_ws, size_t ws_size,
                              hipStream_t stream) {
  (void)in_sizes; (void)n_in; (void)out_size; (void)ws_size;
  const int Bz = 2, T = 2048, D = 2048, F = 8192;
  const int M = Bz * T;  // 4096

  const float* x     = (const float*)d_in[0];
  const float* ln1s  = (const float*)d_in[1];
  const float* ln1b  = (const float*)d_in[2];
  const float* w_qkv = (const float*)d_in[3];
  const float* b_qkv = (const float*)d_in[4];
  const float* ln2s  = (const float*)d_in[5];
  const float* ln2b  = (const float*)d_in[6];
  const float* w1    = (const float*)d_in[7];
  const float* b1    = (const float*)d_in[8];
  const float* w2    = (const float*)d_in[9];
  const float* b2    = (const float*)d_in[10];
  float* outp = (float*)d_out;

  // workspace carve (aliased buffers; sequential stream ordering makes it safe)
  char* p = (char*)d_ws;
  bf16* wT   = (bf16*)p;                       // 33.5MB, reused for all 3 weights
  bf16* qkvh = (bf16*)(p + 33554432);          // 67MB: qkv bf16, later h bf16
  bf16* r12  = (bf16*)(p + 33554432 + 67108864);            // 16.8MB ln output
  float* att = (float*)(p + 33554432 + 67108864 + 16777216); // 33.5MB attn out

  dim3 blk256(256), blkT(32, 8);

  // 1) LN1 -> bf16
  ln_kernel<false><<<M, blk256, 0, stream>>>(x, nullptr, ln1s, ln1b, nullptr, r12);
  // 2) QKV GEMM
  convt_kernel<<<dim3(3 * D / 32, D / 32), blkT, 0, stream>>>(w_qkv, wT, D, 3 * D);
  gemm_bf16_wmma<0><<<dim3(3 * D / 128, M / 128), blk256, 0, stream>>>(
      r12, wT, b_qkv, qkvh, nullptr, nullptr, M, 3 * D, D);
  // 3) causal flash attention
  attn_kernel<<<dim3(T / 128, Bz * 16), blk256, 0, stream>>>(qkvh, att);
  // 4) residual + LN2 (x1 -> d_out, normalized bf16 -> r12)
  ln_kernel<true><<<M, blk256, 0, stream>>>(x, att, ln2s, ln2b, outp, r12);
  // 5) MLP up + GELU (h aliases qkv buffer)
  convt_kernel<<<dim3(F / 32, D / 32), blkT, 0, stream>>>(w1, wT, D, F);
  gemm_bf16_wmma<1><<<dim3(F / 128, M / 128), blk256, 0, stream>>>(
      r12, wT, b1, qkvh, nullptr, nullptr, M, F, D);
  // 6) MLP down + bias + residual (reads x1 from d_out, writes d_out)
  convt_kernel<<<dim3(D / 32, F / 32), blkT, 0, stream>>>(w2, wT, F, D);
  gemm_bf16_wmma<2><<<dim3(D / 128, M / 128), blk256, 0, stream>>>(
      qkvh, wT, b2, nullptr, outp, outp, M, D, F);
}
